// MultiHeadedAttention_65146063946608
// MI455X (gfx1250) — compile-verified, matrix operand order checked by reading, not run on hardware
//
#include <hip/hip_runtime.h>
#include <hip/hip_bf16.h>
#include <stdint.h>

// Problem constants (from the reference): B=8, S=512, D=2048, H=16, HD=128.
#define B_  8
#define S_  512
#define D_  2048
#define H_  16
#define HD_ 128

typedef __attribute__((ext_vector_type(16))) __bf16 v16bf;
typedef __attribute__((ext_vector_type(8)))  __bf16 bf16x8;
typedef __attribute__((ext_vector_type(4)))  __bf16 bf16x4;
typedef __attribute__((ext_vector_type(8)))  float  v8f;

__device__ __forceinline__ v8f zero_v8f() {
  v8f z;
#pragma unroll
  for (int i = 0; i < 8; ++i) z[i] = 0.f;
  return z;
}

// Build a 16x16x32 bf16 WMMA A (or B^T-style) fragment for this lane from a
// row-major row pointer. Per ISA 7.12.2: lane's 16 values are two contiguous
// 8-element chunks at K = 8*half and K = 16 + 8*half.
__device__ __forceinline__ v16bf load_frag(const __bf16* p, int half) {
  bf16x8 lo = *(const bf16x8*)(p + 8 * half);
  bf16x8 hi = *(const bf16x8*)(p + 16 + 8 * half);
  v16bf f;
#pragma unroll
  for (int i = 0; i < 8; ++i) { f[i] = lo[i]; f[8 + i] = hi[i]; }
  return f;
}

// Async global->LDS copy, 16B per lane (GLOBAL_LOAD_ASYNC_TO_LDS_B128, GVS
// mode: SGPR 64-bit base + per-lane 32-bit byte offset). Tracked by ASYNCcnt.
__device__ __forceinline__ void async_b128(uint32_t lds_byte, uint32_t gbyte,
                                           const void* base) {
  asm volatile("global_load_async_to_lds_b128 %0, %1, %2"
               :
               : "v"(lds_byte), "v"(gbyte), "s"(base)
               : "memory");
}
__device__ __forceinline__ void wait_async0() {
  asm volatile("s_wait_asynccnt 0" ::: "memory");
}
// Low 32 bits of the generic address == LDS byte offset (ISA 10.2 aperture map).
__device__ __forceinline__ uint32_t lds_addr(const void* p) {
  return (uint32_t)(uintptr_t)p;
}

// ---------------------------------------------------------------------------
// Kernel 0: streaming f32 -> bf16 conversion (one float4 per thread).
// ---------------------------------------------------------------------------
__global__ __launch_bounds__(256) void cvt_kernel(const float* __restrict__ src,
                                                  __bf16* __restrict__ dst,
                                                  int n4) {
  int i = blockIdx.x * 256 + threadIdx.x;
  if (i < n4) {
    float4 v = ((const float4*)src)[i];
    bf16x4 o;
    o[0] = (__bf16)v.x; o[1] = (__bf16)v.y;
    o[2] = (__bf16)v.z; o[3] = (__bf16)v.w;
    *(bf16x4*)(dst + (size_t)i * 4) = o;
  }
}

// ---------------------------------------------------------------------------
// Kernel 1: Y = X @ W^T + bias with X, W pre-converted to bf16.
// Block tile 128(M) x 64(N), K-step 32, double-buffered LDS tiles filled by
// GLOBAL_LOAD_ASYNC_TO_LDS_B128 (overlaps with WMMA issue).
// 8 waves; wave w computes rows [16w,16w+16) x all 64 cols (4 WMMA tiles).
// transposed==0: store [B,H,S,HD]; transposed==1: store V^T [B,H,HD,S].
// ---------------------------------------------------------------------------
__device__ __forceinline__ void issue_stage(
    const __bf16* __restrict__ Xb, const __bf16* __restrict__ Wb,
    int bm, int bn, int kb, int tid, uint32_t ldsA, uint32_t ldsB)
{
  // A tile: 128x32 bf16 = 8 KB = 512 x 16B chunks -> 2 per thread.
#pragma unroll
  for (int i = 0; i < 2; ++i) {
    int c = tid + i * 256;
    int row = c >> 2, seg = c & 3;
    uint32_t goff = (uint32_t)(((bm * 128 + row) * D_ + kb + seg * 8) * 2);
    async_b128(ldsA + (uint32_t)(row * 64 + seg * 16), goff, (const void*)Xb);
  }
  // B tile: 64x32 bf16 = 4 KB = 256 x 16B chunks -> 1 per thread.
  {
    int row = tid >> 2, seg = tid & 3;
    uint32_t goff = (uint32_t)(((bn * 64 + row) * D_ + kb + seg * 8) * 2);
    async_b128(ldsB + (uint32_t)(row * 64 + seg * 16), goff, (const void*)Wb);
  }
}

__global__ __launch_bounds__(256) void proj_kernel(
    const __bf16* __restrict__ Xb, const __bf16* __restrict__ Wb,
    const float* __restrict__ bias, __bf16* __restrict__ dst,
    int transposed)
{
  __shared__ alignas(16) __bf16 At[2][128 * 32];   // 2 x 8 KB
  __shared__ alignas(16) __bf16 Bt[2][64 * 32];    // 2 x 4 KB

  const int tid  = threadIdx.x;
  const int lane = tid & 31;
  const int wid  = tid >> 5;
  const int half = lane >> 4;
  const int lrow = lane & 15;
  const int bm   = blockIdx.x;   // 0..31  (M/128)
  const int bn   = blockIdx.y;   // 0..31  (N/64)

  const uint32_t ldsA0 = lds_addr(&At[0][0]);
  const uint32_t ldsB0 = lds_addr(&Bt[0][0]);

  v8f acc[4];
#pragma unroll
  for (int t = 0; t < 4; ++t) acc[t] = zero_v8f();

  issue_stage(Xb, Wb, bm, bn, 0, tid, ldsA0, ldsB0);
  wait_async0();
  __syncthreads();

  const int NSTEP = D_ / 32;
  for (int step = 0; step < NSTEP; ++step) {
    const int cur = step & 1;
    if (step < NSTEP - 1)
      issue_stage(Xb, Wb, bm, bn, (step + 1) * 32, tid,
                  ldsA0 + (uint32_t)((cur ^ 1) * 128 * 32 * 2),
                  ldsB0 + (uint32_t)((cur ^ 1) * 64 * 32 * 2));

    // Batch fragment loads, then issue the 4 WMMAs back-to-back.
    v16bf a = load_frag(&At[cur][(wid * 16 + lrow) * 32], half);
    v16bf bf[4];
#pragma unroll
    for (int t = 0; t < 4; ++t)
      bf[t] = load_frag(&Bt[cur][(t * 16 + lrow) * 32], half);
#pragma unroll
    for (int t = 0; t < 4; ++t)
      acc[t] = __builtin_amdgcn_wmma_f32_16x16x32_bf16(
          false, a, false, bf[t], (short)0, acc[t], false, false);

    if (step < NSTEP - 1) wait_async0();
    __syncthreads();
  }

  // Epilogue: C/D layout (ISA 7.12.2): n = lane&15, m = 8*(lane>>4)+v.
#pragma unroll
  for (int t = 0; t < 4; ++t) {
#pragma unroll
    for (int v = 0; v < 8; ++v) {
      int ml = 8 * half + v;
      int gm = bm * 128 + wid * 16 + ml;       // global row in (B*S)
      int gn = bn * 64 + t * 16 + lrow;        // output channel
      int bb = gm >> 9, ss = gm & (S_ - 1);
      int hh = gn >> 7, hd = gn & (HD_ - 1);
      float val = acc[t][v] + bias[gn];
      size_t off = transposed
          ? ((size_t)(bb * H_ + hh) * HD_ + hd) * S_ + ss
          : ((size_t)(bb * H_ + hh) * S_ + ss) * HD_ + hd;
      dst[off] = (__bf16)val;
    }
  }
}

// ---------------------------------------------------------------------------
// Kernel 2: base[b,q,k] = (1-l1)*l2*time_attn + l1*rel_attn  (head-independent)
// One wave32 per row; 16 elems/lane; shuffle reductions; max-subtracted softmax
// (last row has all-(-10000) rel logits -> must not under/overflow).
// ---------------------------------------------------------------------------
__global__ __launch_bounds__(256) void base_kernel(
    const float* __restrict__ rel, const float* __restrict__ ts,
    const float* __restrict__ l1p, const float* __restrict__ l2p,
    float* __restrict__ base)
{
  const int tid  = threadIdx.x;
  const int lane = tid & 31;
  const int warp = tid >> 5;
  const int ridx = blockIdx.x * 8 + warp;   // 0..B*S-1
  const int b = ridx >> 9;
  const int q = ridx & (S_ - 1);
  const float l1 = l1p[0], l2 = l2p[0];
  const float c1 = (1.f - l1) * l2;

  const float* relRow = rel + ((size_t)b * S_ + q) * S_;
  const float* tsRow  = ts  + ((size_t)b * S_ + q) * S_;

  float xr[16], xt[16];
  float mr = -3.4e38f, mt = -3.4e38f;
#pragma unroll
  for (int i = 0; i < 16; ++i) {
    int k = lane + i * 32;
    float rv = relRow[k];
    float tv = tsRow[k];
    bool fut = (k > q);                              // mask = triu(k>q)
    float a = (fut && rv != 0.f) ? rv : -10000.f;    // relm==0 -> -10000
    float t = fut ? -__builtin_inff() : __expf(-fabsf(tv));
    xr[i] = a; xt[i] = t;
    mr = fmaxf(mr, a); mt = fmaxf(mt, t);
  }
#pragma unroll
  for (int o = 16; o > 0; o >>= 1) {
    mr = fmaxf(mr, __shfl_xor(mr, o, 32));
    mt = fmaxf(mt, __shfl_xor(mt, o, 32));
  }
  float sr = 0.f, st = 0.f;
#pragma unroll
  for (int i = 0; i < 16; ++i) {
    xr[i] = __expf(xr[i] - mr);
    xt[i] = (xt[i] == -__builtin_inff()) ? 0.f : __expf(xt[i] - mt);
    sr += xr[i]; st += xt[i];
  }
#pragma unroll
  for (int o = 16; o > 0; o >>= 1) {
    sr += __shfl_xor(sr, o, 32);
    st += __shfl_xor(st, o, 32);
  }
  const float ir = l1 / sr, it = c1 / st;
  float* outRow = base + ((size_t)b * S_ + q) * S_;
#pragma unroll
  for (int i = 0; i < 16; ++i) {
    int k = lane + i * 32;
    outRow[k] = xr[i] * ir + xt[i] * it;
  }
}

// ---------------------------------------------------------------------------
// Kernel 3: per (b, h, 32-row q-tile):
//   phase 1: scores = Q K^T / sqrt(HD) (+causal mask)  -> LDS f32 [32][512]
//   phase 2: softmax + blend with base; write prob_attn f32; bf16 copy in LDS
//   phase 3: out = P @ V via WMMA against V^T rows
// 2 waves (64 threads); dynamic LDS = 32*512*(4+2) = 96 KB (<= 320 KB/WGP).
// ---------------------------------------------------------------------------
__global__ __launch_bounds__(64) void attn_kernel(
    const __bf16* __restrict__ qb, const __bf16* __restrict__ kbuf,
    const __bf16* __restrict__ vtb, const float* __restrict__ base,
    const float* __restrict__ l1p, const float* __restrict__ l2p,
    float* __restrict__ out, float* __restrict__ prob)
{
  extern __shared__ char smem[];
  float*  sc = (float*)smem;                                   // [32][512]
  __bf16* pb = (__bf16*)(smem + 32 * 512 * sizeof(float));     // [32][512]

  const int tid  = threadIdx.x;
  const int lane = tid & 31;
  const int w    = tid >> 5;          // wave 0/1
  const int half = lane >> 4;
  const int lrow = lane & 15;
  const int qt   = blockIdx.x;        // 0..15
  const int h    = blockIdx.y;
  const int b    = blockIdx.z;
  const int bh   = b * H_ + h;
  const int q0   = qt * 32;

  const float scale = 0.08838834764831845f;  // 1/sqrt(128)
  const float l1 = l1p[0], l2 = l2p[0];
  const float c0 = (1.f - l1) * (1.f - l2);

  // Warm caches for the V^T rows phase 3 will consume (global_prefetch_b8).
#pragma unroll
  for (int ct = 0; ct < 8; ++ct)
    __builtin_prefetch(vtb + ((size_t)bh * HD_ + ct * 16 + lrow) * S_, 0, 3);

  // ---- phase 1: scores ----
  const __bf16* qrow = qb + ((size_t)bh * S_ + (q0 + w * 16 + lrow)) * HD_;
  v16bf aq[4];
#pragma unroll
  for (int d = 0; d < 4; ++d) aq[d] = load_frag(qrow + d * 32, half);

  for (int kt = 0; kt < 32; ++kt) {
    v8f acc = zero_v8f();
    const __bf16* krow = kbuf + ((size_t)bh * S_ + (kt * 16 + lrow)) * HD_;
    v16bf bk[4];
#pragma unroll
    for (int d = 0; d < 4; ++d) bk[d] = load_frag(krow + d * 32, half);
#pragma unroll
    for (int d = 0; d < 4; ++d)
      acc = __builtin_amdgcn_wmma_f32_16x16x32_bf16(
          false, aq[d], false, bk[d], (short)0, acc, false, false);
#pragma unroll
    for (int v = 0; v < 8; ++v) {
      int ml = 8 * half + v;
      int grow = q0 + w * 16 + ml;
      int kc = kt * 16 + lrow;
      float val = (kc > grow) ? -1e9f : acc[v] * scale;
      sc[(w * 16 + ml) * 512 + kc] = val;
    }
  }
  __syncthreads();

  // ---- phase 2: softmax + blend (2 threads per row, partners via shfl) ----
  {
    int r    = tid >> 1;              // 0..31
    int part = tid & 1;
    int grow = q0 + r;
    const float* srow = sc + r * 512;
    const float* brow = base + ((size_t)b * S_ + grow) * S_;
    float*       prow = prob + ((size_t)bh * S_ + grow) * S_;
    int k0 = part * 256;

    float m = -3.4e38f;
    for (int k = k0; k < k0 + 256; ++k) m = fmaxf(m, srow[k]);
    m = fmaxf(m, __shfl_xor(m, 1, 32));
    float s = 0.f;
    for (int k = k0; k < k0 + 256; ++k) s += __expf(srow[k] - m);
    s += __shfl_xor(s, 1, 32);
    float inv = c0 / s;
    for (int k = k0; k < k0 + 256; ++k) {
      float p = __expf(srow[k] - m) * inv + brow[k];
      prow[k] = p;
      pb[r * 512 + k] = (__bf16)p;
    }
  }
  __syncthreads();

  // ---- phase 3: out = P @ V (wave computes 16 x 128 strip) ----
  v8f oacc[8];
#pragma unroll
  for (int ct = 0; ct < 8; ++ct) oacc[ct] = zero_v8f();

  for (int ks = 0; ks < 16; ++ks) {
    v16bf ap = load_frag(&pb[(w * 16 + lrow) * 512 + ks * 32], half);
#pragma unroll
    for (int cg = 0; cg < 2; ++cg) {       // two groups of 4: batch loads, then WMMAs
      v16bf bv[4];
#pragma unroll
      for (int t = 0; t < 4; ++t) {
        int ct = cg * 4 + t;
        const __bf16* vrow =
            vtb + ((size_t)bh * HD_ + (ct * 16 + lrow)) * S_ + ks * 32;
        bv[t] = load_frag(vrow, half);
      }
#pragma unroll
      for (int t = 0; t < 4; ++t) {
        int ct = cg * 4 + t;
        oacc[ct] = __builtin_amdgcn_wmma_f32_16x16x32_bf16(
            false, ap, false, bv[t], (short)0, oacc[ct], false, false);
      }
    }
  }
#pragma unroll
  for (int ct = 0; ct < 8; ++ct) {
#pragma unroll
    for (int v = 0; v < 8; ++v) {
      int ml = 8 * half + v;
      int grow = q0 + w * 16 + ml;
      out[((size_t)b * S_ + grow) * D_ + h * HD_ + ct * 16 + lrow] = oacc[ct][v];
    }
  }
}

// ---------------------------------------------------------------------------
extern "C" void kernel_launch(void* const* d_in, const int* in_sizes, int n_in,
                              void* d_out, int out_size, void* d_ws, size_t ws_size,
                              hipStream_t stream) {
  const float* query = (const float*)d_in[0];
  const float* key   = (const float*)d_in[1];
  const float* value = (const float*)d_in[2];
  const float* rel   = (const float*)d_in[3];
  const float* l1    = (const float*)d_in[4];
  const float* l2    = (const float*)d_in[5];
  const float* ts    = (const float*)d_in[6];
  // d_in[7], d_in[8] (pos embeds) unused by reference; d_in[9] mask is analytic.
  const float* Wq = (const float*)d_in[10]; const float* bq = (const float*)d_in[11];
  const float* Wk = (const float*)d_in[12]; const float* bk = (const float*)d_in[13];
  const float* Wv = (const float*)d_in[14]; const float* bv = (const float*)d_in[15];

  // Workspace layout (80 MB total):
  //   q,k bf16 [B,H,S,HD]; v^T bf16 [B,H,HD,S]; base f32 [B,S,S];
  //   xb bf16 (4096x2048) and wb bf16 (2048x2048) staging, reused per proj.
  char* ws = (char*)d_ws;
  const size_t QKV_BYTES = (size_t)B_ * H_ * S_ * HD_ * sizeof(__bf16); // 16 MB
  __bf16* qbuf = (__bf16*)(ws);
  __bf16* kbuf = (__bf16*)(ws + QKV_BYTES);
  __bf16* vtb  = (__bf16*)(ws + 2 * QKV_BYTES);
  float*  bse  = (float*)(ws + 3 * QKV_BYTES);                          // 8 MB
  __bf16* xb   = (__bf16*)(ws + 3 * QKV_BYTES + (size_t)B_ * S_ * S_ * 4);
  __bf16* wb   = (__bf16*)((char*)xb + (size_t)B_ * S_ * D_ * sizeof(__bf16));

  float* outp  = (float*)d_out;                        // (B,S,D)
  float* probp = (float*)d_out + (size_t)B_ * S_ * D_; // (B,H,S,S)

  const int nX4 = (B_ * S_ * D_) / 4;   // 2,097,152 float4s
  const int nW4 = (D_ * D_) / 4;        // 1,048,576 float4s
  dim3 g1(32, 32, 1), t1(256);

  cvt_kernel<<<dim3(nX4 / 256), t1, 0, stream>>>(query, xb, nX4);
  cvt_kernel<<<dim3(nW4 / 256), t1, 0, stream>>>(Wq, wb, nW4);
  proj_kernel<<<g1, t1, 0, stream>>>(xb, wb, bq, qbuf, 0);

  cvt_kernel<<<dim3(nX4 / 256), t1, 0, stream>>>(key, xb, nX4);
  cvt_kernel<<<dim3(nW4 / 256), t1, 0, stream>>>(Wk, wb, nW4);
  proj_kernel<<<g1, t1, 0, stream>>>(xb, wb, bk, kbuf, 0);

  cvt_kernel<<<dim3(nX4 / 256), t1, 0, stream>>>(value, xb, nX4);
  cvt_kernel<<<dim3(nW4 / 256), t1, 0, stream>>>(Wv, wb, nW4);
  proj_kernel<<<g1, t1, 0, stream>>>(xb, wb, bv, vtb, 1);

  base_kernel<<<dim3((B_ * S_) / 8), dim3(256), 0, stream>>>(rel, ts, l1, l2, bse);

  size_t shmem = 32 * 512 * sizeof(float) + 32 * 512 * sizeof(__bf16); // 96 KB
  attn_kernel<<<dim3(S_ / 32, H_, B_), dim3(64), shmem, stream>>>(
      qbuf, kbuf, vtb, bse, l1, l2, outp, probp);
}
